// CombinedModel_70841190580272
// MI455X (gfx1250) — matrix operand order repaired and checked by reading, not compiled, not run on hardware
//
#include <hip/hip_runtime.h>
#include <hip/hip_bf16.h>

#define N_NODESC 250000
#define N_EDGESC 4000000
#define BGR  512
#define LSEQ 20
#define TSEQ 21
#define DMC  32
#define NHC  4
#define FFC  2048
#define GOC  64

typedef __attribute__((ext_vector_type(16))) _Float16 v16h;
typedef __attribute__((ext_vector_type(8)))  _Float16 v8h;
typedef __attribute__((ext_vector_type(8)))  float    v8f;

// ---------------------------------------------------------------- WMMA core
__device__ __forceinline__ v8f wmma_f16(v16h a, v16h b, v8f c) {
  // D(f32 16x16) = A(f16 16x32) * B(f16 32x16) + C
  return __builtin_amdgcn_wmma_f32_16x16x32_f16(false, a, false, b, (short)0, c,
                                                false, false);
}

// Branch-free fast loaders. Preconditions (guaranteed by callers):
//   M%16==0, K%32==0, lda/ldw%4==0, 16B-aligned bases.
// A layout: lane L holds M=row0+L%16; elements = two contiguous 8-float runs
// at k0+8*hi and k0+16+8*hi (hi = L/16).
__device__ __forceinline__ v16h load_a_fast(const float* __restrict__ A, int lda,
                                            int row0, int k0, int lane) {
  int m = row0 + (lane & 15);
  int hi = (lane >> 4) & 1;
  const float* p = A + (size_t)m * lda + k0 + 8 * hi;
  float4 f0 = *(const float4*)(p + 0);
  float4 f1 = *(const float4*)(p + 4);
  float4 f2 = *(const float4*)(p + 16);
  float4 f3 = *(const float4*)(p + 20);
  v16h a;
  a[0] = (_Float16)f0.x;  a[1] = (_Float16)f0.y;  a[2] = (_Float16)f0.z;  a[3] = (_Float16)f0.w;
  a[4] = (_Float16)f1.x;  a[5] = (_Float16)f1.y;  a[6] = (_Float16)f1.z;  a[7] = (_Float16)f1.w;
  a[8] = (_Float16)f2.x;  a[9] = (_Float16)f2.y;  a[10] = (_Float16)f2.z; a[11] = (_Float16)f2.w;
  a[12] = (_Float16)f3.x; a[13] = (_Float16)f3.y; a[14] = (_Float16)f3.z; a[15] = (_Float16)f3.w;
  return a;
}

// B tile from W stored [N x K] row-major (A @ W^T). Lane L holds N=col0+L%16;
// elements = two contiguous 8-float runs at k0+8*hi and k0+16+8*hi.
__device__ __forceinline__ v16h load_bT_fast(const float* __restrict__ W, int ldw,
                                             int col0, int k0, int lane) {
  int n = col0 + (lane & 15);
  int hi = (lane >> 4) & 1;
  const float* p = W + (size_t)n * ldw + k0 + 8 * hi;
  float4 f0 = *(const float4*)(p + 0);
  float4 f1 = *(const float4*)(p + 4);
  float4 f2 = *(const float4*)(p + 16);
  float4 f3 = *(const float4*)(p + 20);
  v16h b;
  b[0] = (_Float16)f0.x;  b[1] = (_Float16)f0.y;  b[2] = (_Float16)f0.z;  b[3] = (_Float16)f0.w;
  b[4] = (_Float16)f1.x;  b[5] = (_Float16)f1.y;  b[6] = (_Float16)f1.z;  b[7] = (_Float16)f1.w;
  b[8] = (_Float16)f2.x;  b[9] = (_Float16)f2.y;  b[10] = (_Float16)f2.z; b[11] = (_Float16)f2.w;
  b[12] = (_Float16)f3.x; b[13] = (_Float16)f3.y; b[14] = (_Float16)f3.z; b[15] = (_Float16)f3.w;
  return b;
}

// WMMA GEMM: C[M,N] = act(A[M,K] @ W^T + bias (+C if acc)), W stored [N,K].
// One wave computes TN adjacent 16x16 C tiles sharing one A fragment.
// Requires M%16==0, K%32==0, (N/16)%TN==0.
template <int TN>
__global__ void k_gemm_t(const float* __restrict__ A, const float* __restrict__ Bm,
                         const float* __restrict__ bias, float* __restrict__ C,
                         int K, int N, int lda, int ldb, int relu, int acc) {
  int ng = (N >> 4) / TN;
  int tile = blockIdx.x;
  int tm = tile / ng, tg = tile % ng;
  int lane = threadIdx.x;
  v8f c[TN];
#pragma unroll
  for (int t = 0; t < TN; ++t) c[t] = (v8f){};
  for (int k0 = 0; k0 < K; k0 += 32) {
    v16h a = load_a_fast(A, lda, tm * 16, k0, lane);
#pragma unroll
    for (int t = 0; t < TN; ++t) {
      v16h b = load_bT_fast(Bm, ldb, (tg * TN + t) * 16, k0, lane);
      c[t] = wmma_f16(a, b, c[t]);
    }
  }
  int hi = (lane >> 4) & 1;
  int nn = lane & 15;
#pragma unroll
  for (int t = 0; t < TN; ++t) {
    int n = (tg * TN + t) * 16 + nn;
    float bv = bias ? bias[n] : 0.0f;
#pragma unroll
    for (int r = 0; r < 8; ++r) {
      int m = tm * 16 + 8 * hi + r;   // C layout: M = r + 8*(lane/16)
      float v = c[t][r] + bv;
      if (acc) v += C[(size_t)m * N + n];
      if (relu) v = fmaxf(v, 0.0f);
      C[(size_t)m * N + n] = v;
    }
  }
}

// Fused FF block: Y[16 rows,32] = relu(X@W1^T + b1) @ W2^T + b2; hidden (2048)
// streamed in 32-chunks, repacked to the f16 A-layout through 1KB of LDS.
__global__ void k_ff_fused(const float* __restrict__ X, const float* __restrict__ W1,
                           const float* __restrict__ b1, const float* __restrict__ W2,
                           const float* __restrict__ b2, float* __restrict__ Y, int R) {
  __shared__ _Float16 Hs[16][32];
  int row0 = blockIdx.x * 16;
  int lane = threadIdx.x;
  int hi = (lane >> 4) & 1;
  int nn = lane & 15;
  v16h a = load_a_fast(X, DMC, row0, 0, lane);   // 16x32 activations, reused 64x
  v8f acc0 = {}, acc1 = {};
  for (int c0 = 0; c0 < FFC; c0 += 32) {
    // up-project: H[16, c0..c0+31] (two 16x16 tiles)
    v16h bw0 = load_bT_fast(W1, DMC, c0, 0, lane);
    v16h bw1 = load_bT_fast(W1, DMC, c0 + 16, 0, lane);
    v8f h0 = {}, h1 = {};
    h0 = wmma_f16(a, bw0, h0);
    h1 = wmma_f16(a, bw1, h1);
    __syncthreads();
#pragma unroll
    for (int r = 0; r < 8; ++r) {
      int m = 8 * hi + r;
      Hs[m][nn]      = (_Float16)fmaxf(h0[r] + b1[c0 + nn], 0.0f);
      Hs[m][16 + nn] = (_Float16)fmaxf(h1[r] + b1[c0 + 16 + nn], 0.0f);
    }
    __syncthreads();
    // repack hidden tile into A layout: two 16B LDS loads per lane
    v8h lo = *(const v8h*)&Hs[nn][8 * hi];
    v8h hh = *(const v8h*)&Hs[nn][16 + 8 * hi];
    v16h a2;
#pragma unroll
    for (int e = 0; e < 8; ++e) { a2[e] = lo[e]; a2[8 + e] = hh[e]; }
    // down-project: Y[16,32] += H(16x32) @ W2[:, c0..c0+31]^T
    v16h bb0 = load_bT_fast(W2, FFC, 0, c0, lane);
    v16h bb1 = load_bT_fast(W2, FFC, 16, c0, lane);
    acc0 = wmma_f16(a2, bb0, acc0);
    acc1 = wmma_f16(a2, bb1, acc1);
  }
#pragma unroll
  for (int r = 0; r < 8; ++r) {
    int m = row0 + 8 * hi + r;
    if (m < R) {
      Y[(size_t)m * DMC + nn]      = acc0[r] + b2[nn];
      Y[(size_t)m * DMC + 16 + nn] = acc1[r] + b2[16 + nn];
    }
  }
}

// ---------------------------------------------------------------- utility
__global__ void k_fill(float* __restrict__ p, float v, long long n) {
  long long i = (long long)blockIdx.x * blockDim.x + threadIdx.x;
  if (i < n) p[i] = v;
}

// WT[n,k] = W[k,n] (for tiny conv weights, so every GEMM uses the W^T loader)
__global__ void k_transpose(const float* __restrict__ W, float* __restrict__ WT,
                            int K, int N) {
  int idx = blockIdx.x * blockDim.x + threadIdx.x;
  if (idx >= K * N) return;
  int k = idx / N, n = idx % N;
  WT[(size_t)n * K + k] = W[idx];
}

__device__ __forceinline__ void atomicMaxF(float* addr, float val) {
  if (val >= 0.0f) atomicMax((int*)addr, __float_as_int(val));
  else             atomicMin((unsigned int*)addr, (unsigned int)__float_as_int(val));
}

// ---------------------------------------------------------------- GCN
// conv1 feature transform (K=3 -- not a WMMA candidate): out[n,0:32]
__global__ void k_conv1(const float* __restrict__ x, const float* __restrict__ W,
                        float* __restrict__ out) {
  int n = blockIdx.x * blockDim.x + threadIdx.x;
  if (n >= N_NODESC) return;
  float x0 = x[3 * n], x1 = x[3 * n + 1], x2 = x[3 * n + 2];
  float* o = out + (size_t)n * 32;
#pragma unroll
  for (int j = 0; j < 32; ++j)
    o[j] = x0 * W[j] + x1 * W[32 + j] + x2 * W[64 + j];
}

__global__ void k_deg(const int* __restrict__ col, float* __restrict__ deg) {
  int i = blockIdx.x * blockDim.x + threadIdx.x;
  if (i < N_EDGESC) atomicAdd(&deg[col[i]], 1.0f);
}

__global__ void k_rsqrt(float* __restrict__ p, int n) {
  int i = blockIdx.x * blockDim.x + threadIdx.x;
  if (i < n) { float d = p[i]; p[i] = (d > 0.0f) ? rsqrtf(d) : 0.0f; }
}

// out[col] += hw[row] * dinv[row]*dinv[col], float4 per thread
__global__ void k_scatter_edges(const float* __restrict__ hw, float* __restrict__ out,
                                const int* __restrict__ row, const int* __restrict__ col,
                                const float* __restrict__ dinv, int Fd, int shift) {
  long long idx = (long long)blockIdx.x * blockDim.x + threadIdx.x;
  int e = (int)(idx >> shift);
  if (e >= N_EDGESC) return;
  int c4 = ((int)idx & ((1 << shift) - 1)) * 4;
  int r = row[e], cc = col[e];
  float w = dinv[r] * dinv[cc];
  const float* src = hw + (size_t)r * Fd + c4;
  float* dst = out + (size_t)cc * Fd + c4;
  atomicAdd(dst + 0, src[0] * w);
  atomicAdd(dst + 1, src[1] * w);
  atomicAdd(dst + 2, src[2] * w);
  atomicAdd(dst + 3, src[3] * w);
}

// self-loop contribution + bias + relu
__global__ void k_self_bias_relu(float* __restrict__ out, const float* __restrict__ hw,
                                 const float* __restrict__ dinv,
                                 const float* __restrict__ bias, int Fd) {
  long long idx = (long long)blockIdx.x * blockDim.x + threadIdx.x;
  if (idx >= (long long)N_NODESC * Fd) return;
  int n = (int)(idx / Fd);
  int f = (int)(idx % Fd);
  float di = dinv[n];
  float v = out[idx] + hw[idx] * di * di + bias[f];
  out[idx] = fmaxf(v, 0.0f);
}

// ---------------------------------------------------------------- Set2Set
__global__ void k_lstm_gates(const float* __restrict__ g, const float* __restrict__ bhh,
                             float* __restrict__ hB, float* __restrict__ cB) {
  int idx = blockIdx.x * blockDim.x + threadIdx.x;
  if (idx >= BGR * GOC) return;
  int b = idx / GOC, j = idx % GOC;
  const float* gr = g + (size_t)b * 4 * GOC;
  float i_ = 1.0f / (1.0f + expf(-(gr[j] + bhh[j])));
  float f_ = 1.0f / (1.0f + expf(-(gr[GOC + j] + bhh[GOC + j])));
  float gg = tanhf(gr[2 * GOC + j] + bhh[2 * GOC + j]);
  float o_ = 1.0f / (1.0f + expf(-(gr[3 * GOC + j] + bhh[3 * GOC + j])));
  float c = f_ * cB[idx] + i_ * gg;
  cB[idx] = c;
  hB[idx] = o_ * tanhf(c);
}

__global__ void k_att_e(const float* __restrict__ xg, const float* __restrict__ hB,
                        const int* __restrict__ batch, float* __restrict__ e) {
  int n = blockIdx.x * blockDim.x + threadIdx.x;
  if (n >= N_NODESC) return;
  const float* xr = xg + (size_t)n * GOC;
  const float* hr = hB + (size_t)batch[n] * GOC;
  float s = 0.0f;
#pragma unroll 8
  for (int f = 0; f < GOC; ++f) s += xr[f] * hr[f];
  e[n] = s;
}

__global__ void k_segmax(const float* __restrict__ e, const int* __restrict__ batch,
                         float* __restrict__ m) {
  int n = blockIdx.x * blockDim.x + threadIdx.x;
  if (n < N_NODESC) atomicMaxF(&m[batch[n]], e[n]);
}

__global__ void k_expsum(float* __restrict__ e, const int* __restrict__ batch,
                         const float* __restrict__ m, float* __restrict__ ssum) {
  int n = blockIdx.x * blockDim.x + threadIdx.x;
  if (n >= N_NODESC) return;
  int b = batch[n];
  float a = expf(e[n] - m[b]);
  e[n] = a;
  atomicAdd(&ssum[b], a);
}

__global__ void k_rscatter(const float* __restrict__ e, const float* __restrict__ ssum,
                           const float* __restrict__ xg, const int* __restrict__ batch,
                           float* __restrict__ r) {
  long long idx = (long long)blockIdx.x * blockDim.x + threadIdx.x;
  if (idx >= (long long)N_NODESC * GOC) return;
  int n = (int)(idx / GOC), f = (int)(idx % GOC);
  int b = batch[n];
  atomicAdd(&r[(size_t)b * GOC + f], (e[n] / ssum[b]) * xg[(size_t)n * GOC + f]);
}

__global__ void k_qstar(const float* __restrict__ hB, const float* __restrict__ r,
                        float* __restrict__ q) {
  int idx = blockIdx.x * blockDim.x + threadIdx.x;
  if (idx >= BGR * 2 * GOC) return;
  int b = idx / (2 * GOC), j = idx % (2 * GOC);
  q[idx] = (j < GOC) ? hB[(size_t)b * GOC + j] : r[(size_t)b * GOC + (j - GOC)];
}

// ---------------------------------------------------------------- decoder
__global__ void k_embed(const int* __restrict__ tokens, const float* __restrict__ emb,
                        float* __restrict__ tgt) {
  int idx = blockIdx.x * blockDim.x + threadIdx.x;
  if (idx >= BGR * TSEQ * DMC) return;
  int d = idx % DMC;
  int t = (idx / DMC) % TSEQ;
  int b = idx / (DMC * TSEQ);
  int tok = (t == 0) ? 0 : (tokens[b * LSEQ + (t - 1)] + 1);
  float div = expf(-logf(10000.0f) * (float)(d & ~1) / (float)DMC);
  float pe = (d & 1) ? cosf((float)t * div) : sinf((float)t * div);
  tgt[idx] = emb[tok * DMC + d] + pe;
}

__global__ void k_scores(const float* __restrict__ qkv, float* __restrict__ s) {
  int idx = blockIdx.x * blockDim.x + threadIdx.x;
  if (idx >= BGR * NHC * TSEQ * TSEQ) return;
  int ki = idx % TSEQ;
  int rest = idx / TSEQ;
  int qi = rest % TSEQ; rest /= TSEQ;
  int h = rest % NHC; int b = rest / NHC;
  const float* q = qkv + (size_t)(b * TSEQ + qi) * 96 + h * 8;
  const float* k = qkv + (size_t)(b * TSEQ + ki) * 96 + DMC + h * 8;
  float acc = 0.0f;
#pragma unroll
  for (int d = 0; d < 8; ++d) acc += q[d] * k[d];
  acc *= 0.35355339059f;        // 1/sqrt(8)
  if (ki > qi) acc -= 1e9f;     // causal mask
  s[idx] = acc;
}

__global__ void k_softmax(float* __restrict__ s) {
  int r = blockIdx.x * blockDim.x + threadIdx.x;
  if (r >= BGR * NHC * TSEQ) return;
  float* p = s + (size_t)r * TSEQ;
  float mx = p[0];
  for (int i = 1; i < TSEQ; ++i) mx = fmaxf(mx, p[i]);
  float sum = 0.0f;
  for (int i = 0; i < TSEQ; ++i) { float e = expf(p[i] - mx); p[i] = e; sum += e; }
  float inv = 1.0f / sum;
  for (int i = 0; i < TSEQ; ++i) p[i] *= inv;
}

__global__ void k_attv(const float* __restrict__ qkv, const float* __restrict__ s,
                       float* __restrict__ o) {
  int idx = blockIdx.x * blockDim.x + threadIdx.x;
  if (idx >= BGR * TSEQ * DMC) return;
  int colv = idx % DMC; int rowq = idx / DMC;
  int b = rowq / TSEQ, qi = rowq % TSEQ;
  int h = colv / 8, d = colv % 8;
  const float* arow = s + ((size_t)(b * NHC + h) * TSEQ + qi) * TSEQ;
  const float* vcol = qkv + (size_t)b * TSEQ * 96 + 2 * DMC + h * 8 + d;
  float acc = 0.0f;
  for (int ki = 0; ki < TSEQ; ++ki) acc += arow[ki] * vcol[(size_t)ki * 96];
  o[idx] = acc;
}

// x = LN(x + delta); bcast=1: delta is [B,32] broadcast over T
__global__ void k_add_ln(float* __restrict__ x, const float* __restrict__ delta,
                         const float* __restrict__ g, const float* __restrict__ b,
                         int R, int bcast) {
  int rowi = blockIdx.x * blockDim.x + threadIdx.x;
  if (rowi >= R) return;
  float* xr = x + (size_t)rowi * DMC;
  const float* dr = delta + (size_t)(bcast ? (rowi / TSEQ) : rowi) * DMC;
  float t[DMC];
  float mean = 0.0f;
#pragma unroll
  for (int d = 0; d < DMC; ++d) { t[d] = xr[d] + dr[d]; mean += t[d]; }
  mean *= (1.0f / DMC);
  float var = 0.0f;
#pragma unroll
  for (int d = 0; d < DMC; ++d) { float u = t[d] - mean; var += u * u; }
  var *= (1.0f / DMC);
  float inv = rsqrtf(var + 1e-5f);
#pragma unroll
  for (int d = 0; d < DMC; ++d) xr[d] = (t[d] - mean) * inv * g[d] + b[d];
}

__global__ void k_qor(const float* __restrict__ tgt, const float* __restrict__ qw,
                      const float* __restrict__ qb, float* __restrict__ out) {
  int idx = blockIdx.x * blockDim.x + threadIdx.x;
  if (idx >= BGR * LSEQ) return;
  int b = idx / LSEQ, t = idx % LSEQ + 1;   // skip graph token
  const float* xr = tgt + ((size_t)b * TSEQ + t) * DMC;
  float s = qb[0];
#pragma unroll
  for (int d = 0; d < DMC; ++d) s += xr[d] * qw[d];
  out[idx] = s;
}

// ---------------------------------------------------------------- launcher
extern "C" void kernel_launch(void* const* d_in, const int* in_sizes, int n_in,
                              void* d_out, int out_size, void* d_ws, size_t ws_size,
                              hipStream_t stream) {
  auto F = [&](int i) { return (const float*)d_in[i]; };
  const float* x      = F(0);
  const int* row      = (const int*)d_in[1];
  const int* col      = row + N_EDGESC;
  const int* batch    = (const int*)d_in[2];
  const int* tokens   = (const int*)d_in[3];
  const float* conv1_w = F(4),  *conv1_b = F(5);
  const float* conv2_w = F(6),  *conv2_b = F(7);
  const float* conv3_w = F(8),  *conv3_b = F(9);
  const float* lstm_wih = F(10), *lstm_whh = F(11);
  const float* lstm_bih = F(12), *lstm_bhh = F(13);
  const float* proj_w = F(14), *proj_b = F(15);
  const float* emb    = F(16);
  const float* qor_w  = F(17 + 18 * 4), *qor_b = F(17 + 18 * 4 + 1);

  // workspace carve (floats)
  float* w = (float*)d_ws;
  float* bufA    = w; w += (size_t)N_NODESC * 64;
  float* bufB    = w; w += (size_t)N_NODESC * 64;
  float* dinv    = w; w += N_NODESC;
  float* ebuf    = w; w += N_NODESC;
  float* gbuf    = w; w += BGR * 256;
  float* hB      = w; w += BGR * GOC;
  float* cB      = w; w += BGR * GOC;
  float* qstar   = w; w += BGR * 2 * GOC;
  float* mmax    = w; w += BGR;
  float* ssum    = w; w += BGR;
  float* rbuf    = w; w += BGR * GOC;
  float* hgraph  = w; w += BGR * DMC;
  float* tgt     = w; w += BGR * TSEQ * DMC;
  float* qkv     = w; w += BGR * TSEQ * 96;
  float* scor    = w; w += (size_t)BGR * NHC * TSEQ * TSEQ;
  float* obuf    = w; w += BGR * TSEQ * DMC;
  float* sabuf   = w; w += BGR * TSEQ * DMC;
  float* vg      = w; w += BGR * DMC;
  float* cag     = w; w += BGR * DMC;
  float* conv2_wt = w; w += 32 * 64;   // conv2_w^T  [64,32]
  float* conv3_wt = w; w += 64 * 64;   // conv3_w^T  [64,64]

  const int TB = 256;
  auto nb = [&](long long n) { return (unsigned)((n + TB - 1) / TB); };
  // dispatch: every GEMM is A @ W^T with W stored [N,K]; TN tiles per wave.
  auto gemm = [&](const float* A, const float* Bm, const float* bias, float* C,
                  int M, int K, int N, int lda, int ldb, int relu, int acc) {
    int nt = N >> 4;                       // N in {32,64,96,256} -> nt even
    int TN = (nt % 4 == 0) ? 4 : 2;
    dim3 g((unsigned)((M >> 4) * (nt / TN))), b(32);
    if (TN == 4) k_gemm_t<4><<<g, b, 0, stream>>>(A, Bm, bias, C, K, N, lda, ldb, relu, acc);
    else         k_gemm_t<2><<<g, b, 0, stream>>>(A, Bm, bias, C, K, N, lda, ldb, relu, acc);
  };

  // ---- one-time tiny weight transposes (so all B loads are vectorized) ----
  k_transpose<<<nb(32 * 64), TB, 0, stream>>>(conv2_w, conv2_wt, 32, 64);
  k_transpose<<<nb(64 * 64), TB, 0, stream>>>(conv3_w, conv3_wt, 64, 64);

  // ---- GCN normalization (deg includes self loop) ----
  k_fill<<<nb(N_NODESC), TB, 0, stream>>>(dinv, 1.0f, N_NODESC);
  k_deg<<<nb(N_EDGESC), TB, 0, stream>>>(col, dinv);
  k_rsqrt<<<nb(N_NODESC), TB, 0, stream>>>(dinv, N_NODESC);

  // ---- conv1: 3 -> 32 (scalar transform; K=3 is no WMMA candidate) ----
  k_conv1<<<nb(N_NODESC), TB, 0, stream>>>(x, conv1_w, bufA);
  k_fill<<<nb((long long)N_NODESC * 32), TB, 0, stream>>>(bufB, 0.0f, (long long)N_NODESC * 32);
  k_scatter_edges<<<nb((long long)N_EDGESC * 8), TB, 0, stream>>>(bufA, bufB, row, col, dinv, 32, 3);
  k_self_bias_relu<<<nb((long long)N_NODESC * 32), TB, 0, stream>>>(bufB, bufA, dinv, conv1_b, 32);
  // ---- conv2: 32 -> 64 ----
  gemm(bufB, conv2_wt, nullptr, bufA, N_NODESC, 32, 64, 32, 32, 0, 0);
  k_fill<<<nb((long long)N_NODESC * 64), TB, 0, stream>>>(bufB, 0.0f, (long long)N_NODESC * 64);
  k_scatter_edges<<<nb((long long)N_EDGESC * 16), TB, 0, stream>>>(bufA, bufB, row, col, dinv, 64, 4);
  k_self_bias_relu<<<nb((long long)N_NODESC * 64), TB, 0, stream>>>(bufB, bufA, dinv, conv2_b, 64);
  // ---- conv3: 64 -> 64 ----
  gemm(bufB, conv3_wt, nullptr, bufA, N_NODESC, 64, 64, 64, 64, 0, 0);
  k_fill<<<nb((long long)N_NODESC * 64), TB, 0, stream>>>(bufB, 0.0f, (long long)N_NODESC * 64);
  k_scatter_edges<<<nb((long long)N_EDGESC * 16), TB, 0, stream>>>(bufA, bufB, row, col, dinv, 64, 4);
  k_self_bias_relu<<<nb((long long)N_NODESC * 64), TB, 0, stream>>>(bufB, bufA, dinv, conv3_b, 64);
  // xg = bufB

  // ---- Set2Set pooling ----
  k_fill<<<nb(BGR * GOC), TB, 0, stream>>>(hB, 0.0f, BGR * GOC);
  k_fill<<<nb(BGR * GOC), TB, 0, stream>>>(cB, 0.0f, BGR * GOC);
  k_fill<<<nb(BGR * 2 * GOC), TB, 0, stream>>>(qstar, 0.0f, BGR * 2 * GOC);
  for (int it = 0; it < 3; ++it) {
    gemm(qstar, lstm_wih, lstm_bih, gbuf, BGR, 128, 256, 128, 128, 0, 0);
    gemm(hB, lstm_whh, nullptr, gbuf, BGR, 64, 256, 64, 64, 0, 1);
    k_lstm_gates<<<nb(BGR * GOC), TB, 0, stream>>>(gbuf, lstm_bhh, hB, cB);
    k_att_e<<<nb(N_NODESC), TB, 0, stream>>>(bufB, hB, batch, ebuf);
    k_fill<<<nb(BGR), TB, 0, stream>>>(mmax, -1e30f, BGR);
    k_segmax<<<nb(N_NODESC), TB, 0, stream>>>(ebuf, batch, mmax);
    k_fill<<<nb(BGR), TB, 0, stream>>>(ssum, 0.0f, BGR);
    k_expsum<<<nb(N_NODESC), TB, 0, stream>>>(ebuf, batch, mmax, ssum);
    k_fill<<<nb(BGR * GOC), TB, 0, stream>>>(rbuf, 0.0f, BGR * GOC);
    k_rscatter<<<nb((long long)N_NODESC * GOC), TB, 0, stream>>>(ebuf, ssum, bufB, batch, rbuf);
    k_qstar<<<nb(BGR * 2 * GOC), TB, 0, stream>>>(hB, rbuf, qstar);
  }
  gemm(qstar, proj_w, proj_b, hgraph, BGR, 128, 32, 128, 128, 0, 0);

  // ---- Transformer decoder ----
  k_embed<<<nb(BGR * TSEQ * DMC), TB, 0, stream>>>(tokens, emb, tgt);
  const int R = BGR * TSEQ;  // 10752 rows (multiple of 16)
  for (int l = 0; l < 4; ++l) {
    int base = 17 + 18 * l;
    const float* sa_in_w = F(base + 0),  *sa_in_b = F(base + 1);
    const float* sa_out_w = F(base + 2), *sa_out_b = F(base + 3);
    const float* ca_in_w = F(base + 4),  *ca_in_b = F(base + 5);
    const float* ca_out_w = F(base + 6), *ca_out_b = F(base + 7);
    const float* ln1_g = F(base + 8),  *ln1_b = F(base + 9);
    const float* ln2_g = F(base + 10), *ln2_b = F(base + 11);
    const float* ln3_g = F(base + 12), *ln3_b = F(base + 13);
    const float* ff1_w = F(base + 14), *ff1_b = F(base + 15);
    const float* ff2_w = F(base + 16), *ff2_b = F(base + 17);

    // self-attention
    gemm(tgt, sa_in_w, sa_in_b, qkv, R, 32, 96, 32, 32, 0, 0);
    k_scores<<<nb((long long)BGR * NHC * TSEQ * TSEQ), TB, 0, stream>>>(qkv, scor);
    k_softmax<<<nb(BGR * NHC * TSEQ), TB, 0, stream>>>(scor);
    k_attv<<<nb((long long)R * DMC), TB, 0, stream>>>(qkv, scor, obuf);
    gemm(obuf, sa_out_w, sa_out_b, sabuf, R, 32, 32, 32, 32, 0, 0);
    k_add_ln<<<nb(R), TB, 0, stream>>>(tgt, sabuf, ln1_g, ln1_b, R, 0);

    // cross-attention: all memory rows identical => softmax uniform => output
    // collapses to (h_graph @ Wv^T + bv) @ Wo^T + bo, broadcast over T.
    gemm(hgraph, ca_in_w + 64 * 32, ca_in_b + 64, vg, BGR, 32, 32, 32, 32, 0, 0);
    gemm(vg, ca_out_w, ca_out_b, cag, BGR, 32, 32, 32, 32, 0, 0);
    k_add_ln<<<nb(R), TB, 0, stream>>>(tgt, cag, ln2_g, ln2_b, R, 1);

    // fused feed-forward (hidden 2048 never leaves LDS/registers)
    k_ff_fused<<<dim3(R / 16), dim3(32), 0, stream>>>(tgt, ff1_w, ff1_b, ff2_w, ff2_b, sabuf, R);
    k_add_ln<<<nb(R), TB, 0, stream>>>(tgt, sabuf, ln3_g, ln3_b, R, 0);
  }

  k_qor<<<nb(BGR * LSEQ), TB, 0, stream>>>(tgt, qor_w, qor_b, (float*)d_out);
}
